// MultiHeadFastSelfAttention_8959301779873
// MI455X (gfx1250) — matrix operand order repaired and checked by reading, not compiled, not run on hardware
//
#include <hip/hip_runtime.h>

// ---------------------------------------------------------------------------
// MultiHeadFastSelfAttention for MI455X (gfx1250, wave32, WMMA bf16)
// B=4, S=8192, D=1024, H=16, DH=64
// ---------------------------------------------------------------------------

typedef unsigned short u16;
typedef unsigned int   u32;

typedef __attribute__((ext_vector_type(4)))  float  f32x4;
typedef __attribute__((ext_vector_type(4)))  u32    u32x4;
typedef __attribute__((ext_vector_type(2)))  u32    u32x2;
typedef __attribute__((ext_vector_type(16))) __bf16 v16bf;
typedef __attribute__((ext_vector_type(8)))  float  v8f;

struct BFrag { u32x4 lo, hi; };   // 32 bytes == v16bf

static constexpr int Bsz = 4;
static constexpr int S   = 8192;
static constexpr int D   = 1024;
static constexpr int H   = 16;
static constexpr int DH  = 64;
static constexpr int M   = Bsz * S;      // 32768 GEMM rows
static constexpr int LDA = 40;           // LDS row stride (halfwords); 80B = 16B-aligned rows

__device__ __forceinline__ u32 pack2bf(float a, float b) {
  u32 ua = __float_as_uint(a);
  u32 ub = __float_as_uint(b);
  ua = (ua + 0x7FFFu + ((ua >> 16) & 1u)) >> 16;   // RNE
  ub = (ub + 0x7FFFu + ((ub >> 16) & 1u)) >> 16;
  return (ua & 0xFFFFu) | (ub << 16);
}

// ---------------------------------------------------------------------------
// fp32 -> bf16 conversion (4 elems / thread)
// ---------------------------------------------------------------------------
__global__ void cvt_bf16_kernel(const float* __restrict__ src, u16* __restrict__ dst, int n4) {
  int i = blockIdx.x * blockDim.x + threadIdx.x;
  if (i >= n4) return;
  f32x4 v = *(const f32x4*)(src + (size_t)i * 4);
  u32x2 r;
  r.x = pack2bf(v[0], v[1]);
  r.y = pack2bf(v[2], v[3]);
  *(u32x2*)(dst + (size_t)i * 4) = r;
}

// ---------------------------------------------------------------------------
// 128x128x32-tiled bf16 WMMA GEMM:  out[M,N] = A[M,K] * Bw[K,N] + bias (+res)
//  - Ab   : bf16 A  (used when Af == nullptr)
//  - Af   : fp32 A, scaled per-column by aScale[b*D + k] (fused "weighted")
//  - residual: optional fp32 [M,N] added in epilogue
// 8 waves: 2(M) x 4(N); each wave -> 4x2 grid of 16x16 WMMA tiles.
// ---------------------------------------------------------------------------
__global__ __launch_bounds__(256) void gemm_bf16_kernel(
    const u16*  __restrict__ Ab,
    const float* __restrict__ Af,
    const float* __restrict__ aScale,
    const u16*  __restrict__ Bw,
    const float* __restrict__ bias,
    const float* __restrict__ residual,
    float* __restrict__ out)
{
  constexpr int K = 1024, N = 1024;
  __shared__ u16 As[128 * LDA];   // [m][k]
  __shared__ u16 Bs[128 * LDA];   // [n][k]  (transposed tile)

  const int tid   = threadIdx.x;
  const int lane  = tid & 31;
  const int wid   = tid >> 5;
  const int waveM = wid >> 2;     // 0..1
  const int waveN = wid & 3;      // 0..3
  const int hl    = lane >> 4;    // half of wave
  const int l16   = lane & 15;

  const int mBase = blockIdx.y * 128;
  const int nBase = blockIdx.x * 128;

  const float* sc = (Af != nullptr) ? (aScale + (size_t)(mBase >> 13) * D) : nullptr;

  v8f acc[4][2];
#pragma unroll
  for (int i = 0; i < 4; ++i)
#pragma unroll
    for (int j = 0; j < 2; ++j)
      acc[i][j] = v8f{0.f, 0.f, 0.f, 0.f, 0.f, 0.f, 0.f, 0.f};

  for (int k0 = 0; k0 < K; k0 += 32) {
    // -------- stage A (128x32) and B (32x128 -> [n][k]) into LDS ----------
#pragma unroll
    for (int p = 0; p < 2; ++p) {
      const int chunk = p * 256 + tid;           // 512 chunks of 8 elements
      // A: row-major
      {
        const int row = chunk >> 2;              // 0..127
        const int c8  = (chunk & 3) * 8;         // 0,8,16,24
        u32* dstA = (u32*)&As[row * LDA + c8];
        if (Af) {
          const float* g  = Af + (size_t)(mBase + row) * K + k0 + c8;
          const float* s2 = sc + k0 + c8;
          f32x4 v0 = *(const f32x4*)(g);
          f32x4 v1 = *(const f32x4*)(g + 4);
          f32x4 q0 = *(const f32x4*)(s2);
          f32x4 q1 = *(const f32x4*)(s2 + 4);
          dstA[0] = pack2bf(v0[0] * q0[0], v0[1] * q0[1]);
          dstA[1] = pack2bf(v0[2] * q0[2], v0[3] * q0[3]);
          dstA[2] = pack2bf(v1[0] * q1[0], v1[1] * q1[1]);
          dstA[3] = pack2bf(v1[2] * q1[2], v1[3] * q1[3]);
        } else {
          u32x4 v = *(const u32x4*)(Ab + (size_t)(mBase + row) * K + k0 + c8);
          dstA[0] = v.x; dstA[1] = v.y; dstA[2] = v.z; dstA[3] = v.w;
        }
      }
      // B: read [k][n], write transposed [n][k]
      {
        const int kk = chunk >> 4;               // 0..31
        const int n8 = (chunk & 15) * 8;         // 0..120
        u32x4 v = *(const u32x4*)(Bw + (size_t)(k0 + kk) * N + nBase + n8);
        u32 vv[4] = {v.x, v.y, v.z, v.w};
#pragma unroll
        for (int j = 0; j < 4; ++j) {
          Bs[(n8 + 2 * j + 0) * LDA + kk] = (u16)(vv[j] & 0xFFFFu);
          Bs[(n8 + 2 * j + 1) * LDA + kk] = (u16)(vv[j] >> 16);
        }
      }
    }
    __syncthreads();

    // -------- load fragments in ISA lane layout ---------------------------
    // A lane (l16,hl): row m=l16, elements e<8 -> k=hl*8+e ; e>=8 -> k=hl*8+16+e-8
    v16bf afr[4], bfr[2];
#pragma unroll
    for (int ms = 0; ms < 4; ++ms) {
      const int m = waveM * 64 + ms * 16 + l16;
      const u16* p = &As[m * LDA + hl * 8];
      BFrag t;
      t.lo = *(const u32x4*)(p);
      t.hi = *(const u32x4*)(p + 16);
      afr[ms] = __builtin_bit_cast(v16bf, t);
    }
    // B lane: n=l16, elements e<8 -> k=hl*8+e ; e>=8 -> k=hl*8+16+e-8
#pragma unroll
    for (int ns = 0; ns < 2; ++ns) {
      const int n = waveN * 32 + ns * 16 + l16;
      const u16* p = &Bs[n * LDA + hl * 8];
      BFrag t;
      t.lo = *(const u32x4*)(p);
      t.hi = *(const u32x4*)(p + 16);
      bfr[ns] = __builtin_bit_cast(v16bf, t);
    }

#pragma unroll
    for (int ms = 0; ms < 4; ++ms)
#pragma unroll
      for (int ns = 0; ns < 2; ++ns)
        acc[ms][ns] = __builtin_amdgcn_wmma_f32_16x16x32_bf16(
            false, afr[ms], false, bfr[ns], (short)0, acc[ms][ns], false, false);
    __syncthreads();
  }

  // -------- epilogue: D layout -> (m = r + hl*8, n = l16) -----------------
#pragma unroll
  for (int ms = 0; ms < 4; ++ms) {
#pragma unroll
    for (int ns = 0; ns < 2; ++ns) {
      const int n  = nBase + waveN * 32 + ns * 16 + l16;
      const float bn = bias[n];
#pragma unroll
      for (int r = 0; r < 8; ++r) {
        const int m = mBase + waveM * 64 + ms * 16 + hl * 8 + r;
        const size_t idx = (size_t)m * N + n;
        float v = acc[ms][ns][r] + bn;
        if (residual) v += residual[idx];
        out[idx] = v;
      }
    }
  }
}

// ---------------------------------------------------------------------------
// score[b,h,s] = (sum_d src[b,s,d]*(pv?pv[b,d]:1)*Wa[d,h] + ba[h])*0.125 + mask
// 16 rows x 16 heads per 256-thread block.
// ---------------------------------------------------------------------------
__global__ __launch_bounds__(256) void score_kernel(
    const float* __restrict__ src, const float* __restrict__ pv,
    const float* __restrict__ Wa,  const float* __restrict__ ba,
    const float* __restrict__ mask, float* __restrict__ out)
{
  const int tid = threadIdx.x;
  const int h   = tid & 15;
  const int r   = tid >> 4;
  const int row = blockIdx.x * 16 + r;       // 0..32767
  const int b   = row >> 13;
  const int s   = row & (S - 1);
  const float* x = src + (size_t)row * D;
  const float* p = pv ? (pv + (size_t)b * D) : nullptr;

  float acc = ba[h];
  for (int d = 0; d < D; d += 4) {
    f32x4 v = *(const f32x4*)(x + d);
    if (p) {
      f32x4 q = *(const f32x4*)(p + d);
      v[0] *= q[0]; v[1] *= q[1]; v[2] *= q[2]; v[3] *= q[3];
    }
    acc += v[0] * Wa[(d + 0) * H + h] + v[1] * Wa[(d + 1) * H + h]
         + v[2] * Wa[(d + 2) * H + h] + v[3] * Wa[(d + 3) * H + h];
  }
  out[((size_t)(b * H + h) << 13) + s] = acc * 0.125f + mask[((size_t)b << 13) + s];
}

// ---------------------------------------------------------------------------
// One block per (b,h): softmax over S=8192, then
// pooled[b, h*64+d] = (sum_s w[s]*src[b,s,h*64+d]) * (pv?pv[b,h*64+d]:1)
// ---------------------------------------------------------------------------
__global__ __launch_bounds__(256) void softmax_pool_kernel(
    const float* __restrict__ score, const float* __restrict__ src,
    const float* __restrict__ pv, float* __restrict__ pooled)
{
  __shared__ float w[S];
  __shared__ float red[256];
  const int tid = threadIdx.x;
  const int b   = blockIdx.x >> 4;
  const int h   = blockIdx.x & 15;
  const float* row = score + ((size_t)(b * H + h) << 13);

  // max
  float m = -3.402823466e38f;
  for (int s = tid; s < S; s += 256) m = fmaxf(m, row[s]);
  red[tid] = m; __syncthreads();
  for (int off = 128; off > 0; off >>= 1) {
    if (tid < off) red[tid] = fmaxf(red[tid], red[tid + off]);
    __syncthreads();
  }
  m = red[0]; __syncthreads();

  // exp + sum (weights cached in LDS)
  float sum = 0.f;
  for (int s = tid; s < S; s += 256) { float e = __expf(row[s] - m); w[s] = e; sum += e; }
  red[tid] = sum; __syncthreads();
  for (int off = 128; off > 0; off >>= 1) {
    if (tid < off) red[tid] += red[tid + off];
    __syncthreads();
  }
  const float inv = 1.f / red[0];
  __syncthreads();

  // pooling: 4 s-groups x 64 d
  const int sg = tid >> 6, d = tid & 63;
  const int dc = h * DH + d;
  const float sc2 = pv ? pv[(size_t)b * D + dc] : 1.f;
  float acc = 0.f;
  const float* base = src + ((size_t)b * S) * D + dc;
  for (int s = sg; s < S; s += 4) acc += w[s] * base[(size_t)s * D];
  red[tid] = acc; __syncthreads();
  if (sg == 0) {
    float t = red[d] + red[64 + d] + red[128 + d] + red[192 + d];
    pooled[(size_t)b * D + dc] = t * inv * sc2;
  }
}

// ---------------------------------------------------------------------------
// Host launcher
// ---------------------------------------------------------------------------
extern "C" void kernel_launch(void* const* d_in, const int* in_sizes, int n_in,
                              void* d_out, int out_size, void* d_ws, size_t ws_size,
                              hipStream_t stream) {
  const float* hs   = (const float*)d_in[0];
  const float* mask = (const float*)d_in[1];
  const float* Wq   = (const float*)d_in[2];
  const float* bq   = (const float*)d_in[3];
  const float* Wqa  = (const float*)d_in[4];
  const float* bqa  = (const float*)d_in[5];
  const float* Wk   = (const float*)d_in[6];
  const float* bk   = (const float*)d_in[7];
  const float* Wka  = (const float*)d_in[8];
  const float* bka  = (const float*)d_in[9];
  const float* Wt   = (const float*)d_in[10];
  const float* bt   = (const float*)d_in[11];

  // workspace layout
  char* w = (char*)d_ws;
  u16* hsb = (u16*)w;            w += (size_t)M * D * 2;        // 64 MB
  u16* wqb = (u16*)w;            w += (size_t)D * D * 2;        //  2 MB
  u16* wkb = (u16*)w;            w += (size_t)D * D * 2;
  u16* wtb = (u16*)w;            w += (size_t)D * D * 2;
  float* mq = (float*)w;         w += (size_t)M * D * 4;        // 128 MB
  float* mk = (float*)w;         w += (size_t)M * D * 4;        // 128 MB
  float* score = (float*)w;      w += (size_t)Bsz * H * S * 4;  //  2 MB
  float* pq = (float*)w;         w += (size_t)Bsz * D * 4;      // 16 KB
  float* pk = (float*)w;         w += (size_t)Bsz * D * 4;

  // 1) convert hidden states + weights to bf16
  cvt_bf16_kernel<<<(M * D / 4 + 255) / 256, 256, 0, stream>>>(hs, hsb, M * D / 4);
  cvt_bf16_kernel<<<(D * D / 4 + 255) / 256, 256, 0, stream>>>(Wq, wqb, D * D / 4);
  cvt_bf16_kernel<<<(D * D / 4 + 255) / 256, 256, 0, stream>>>(Wk, wkb, D * D / 4);
  cvt_bf16_kernel<<<(D * D / 4 + 255) / 256, 256, 0, stream>>>(Wt, wtb, D * D / 4);

  dim3 gGrid(D / 128, M / 128);
  // 2) mixed_q = hs @ Wq + bq ; mixed_k = hs @ Wk + bk
  gemm_bf16_kernel<<<gGrid, 256, 0, stream>>>(hsb, nullptr, nullptr, wqb, bq, nullptr, mq);
  gemm_bf16_kernel<<<gGrid, 256, 0, stream>>>(hsb, nullptr, nullptr, wkb, bk, nullptr, mk);

  // 3) q attention pooling
  score_kernel<<<M / 16, 256, 0, stream>>>(mq, nullptr, Wqa, bqa, mask, score);
  softmax_pool_kernel<<<Bsz * H, 256, 0, stream>>>(score, mq, nullptr, pq);

  // 4) k attention pooling on mixed_qk = mixed_k * pooled_q (fused via pv)
  score_kernel<<<M / 16, 256, 0, stream>>>(mk, pq, Wka, bka, mask, score);
  softmax_pool_kernel<<<Bsz * H, 256, 0, stream>>>(score, mk, pq, pk);

  // 5) out = (q_heads * pooled_k) @ Wt + bt + mixed_q
  //    A-scaling by pooled_k fused in staging; residual fused in epilogue.
  gemm_bf16_kernel<<<gGrid, 256, 0, stream>>>(nullptr, mq, pk, wtb, bt, mq, (float*)d_out);
}
